// FakeNewsModel_29910152249873
// MI455X (gfx1250) — compile-verified, matrix operand order checked by reading, not run on hardware
//
#include <hip/hip_runtime.h>
#include <hip/hip_bf16.h>

// ---------------------------------------------------------------------------
// GraphSAGE (3-layer, mean aggregation) forward for gfx1250 (CDNA5, wave32)
//   x = [xc@Wp+bp | xs@Ws+bs]                 (WMMA f16->f32 GEMM)
//   3x: relu( mean_agg(x)@Wl + bl + x@Wr )    (scatter-add + WMMA GEMM)
//   out = x@Wo + bo                           (scalar head, NCLS=2)
//
// Weights are prepacked once into f16 WMMA B-fragments (register layout per
// ISA 7.12.2) so the GEMM inner loop does one contiguous 32B load per B
// operand. N % 16 == 0 so all surviving 16-row tiles are fully in bounds:
// no clamps, no guarded stores.
// ---------------------------------------------------------------------------

#define NNODES 50000
#define DFEAT  64
#define FRAGH  512   // halves per packed fragment (32 lanes x 16)

typedef __attribute__((ext_vector_type(16))) _Float16 v16h;
typedef __attribute__((ext_vector_type(8)))  float    v8f;

__device__ __forceinline__ v8f wmma_f16(v16h a, v16h b, v8f c) {
  return __builtin_amdgcn_wmma_f32_16x16x32_f16(
      /*neg_a=*/false, a, /*neg_b=*/false, b,
      /*c_mod=*/(short)0, c, /*reuse_a=*/false, /*reuse_b=*/false);
}

// ---- A fragment: rows rbase..rbase+15 of row-major X[ld], k-tile kbase. ----
// Lane m=lane%16 holds row m; element j -> k = (j>=8?16:0) + (lane/16)*8 + j%8.
__device__ __forceinline__ v16h load_a_frag(const float* __restrict__ X, int ld,
                                            int rbase, int kbase, int lane) {
  const int m  = lane & 15;
  const int hi = lane >> 4;
  const float* p = X + (size_t)(rbase + m) * ld + kbase + hi * 8;
  const float4 q0 = *(const float4*)(p + 0);
  const float4 q1 = *(const float4*)(p + 4);
  const float4 q2 = *(const float4*)(p + 16);
  const float4 q3 = *(const float4*)(p + 20);
  v16h a;
  a[0]  = (_Float16)q0.x; a[1]  = (_Float16)q0.y;
  a[2]  = (_Float16)q0.z; a[3]  = (_Float16)q0.w;
  a[4]  = (_Float16)q1.x; a[5]  = (_Float16)q1.y;
  a[6]  = (_Float16)q1.z; a[7]  = (_Float16)q1.w;
  a[8]  = (_Float16)q2.x; a[9]  = (_Float16)q2.y;
  a[10] = (_Float16)q2.z; a[11] = (_Float16)q2.w;
  a[12] = (_Float16)q3.x; a[13] = (_Float16)q3.y;
  a[14] = (_Float16)q3.z; a[15] = (_Float16)q3.w;
  return a;
}

// Row-scaled variant (folds mean-aggregation 1/deg into the A operand).
__device__ __forceinline__ v16h load_a_frag_scaled(const float* __restrict__ X,
                                                   int ld, int rbase, int kbase,
                                                   int lane,
                                                   const float* __restrict__ s) {
  const int m  = lane & 15;
  const int hi = lane >> 4;
  const int row = rbase + m;
  const float sc = s[row];
  const float* p = X + (size_t)row * ld + kbase + hi * 8;
  const float4 q0 = *(const float4*)(p + 0);
  const float4 q1 = *(const float4*)(p + 4);
  const float4 q2 = *(const float4*)(p + 16);
  const float4 q3 = *(const float4*)(p + 20);
  v16h a;
  a[0]  = (_Float16)(q0.x * sc); a[1]  = (_Float16)(q0.y * sc);
  a[2]  = (_Float16)(q0.z * sc); a[3]  = (_Float16)(q0.w * sc);
  a[4]  = (_Float16)(q1.x * sc); a[5]  = (_Float16)(q1.y * sc);
  a[6]  = (_Float16)(q1.z * sc); a[7]  = (_Float16)(q1.w * sc);
  a[8]  = (_Float16)(q2.x * sc); a[9]  = (_Float16)(q2.y * sc);
  a[10] = (_Float16)(q2.z * sc); a[11] = (_Float16)(q2.w * sc);
  a[12] = (_Float16)(q3.x * sc); a[13] = (_Float16)(q3.y * sc);
  a[14] = (_Float16)(q3.z * sc); a[15] = (_Float16)(q3.w * sc);
  return a;
}

// ---- Packed B fragment: one contiguous 32B per lane. -----------------------
__device__ __forceinline__ v16h load_b_packed(const _Float16* __restrict__ P,
                                              int nt, int ki, int ni, int lane) {
  return *((const v16h*)P + ((ki * nt + ni) * 32 + lane));
}

// Pack f32 W[K,ncols] into f16 B-fragments (ISA 7.12.2 16-bit B layout).
// One thread per (fragment, lane): writes 16 contiguous halves.
__global__ void pack_b_kernel(const float* __restrict__ W, int K, int ncols,
                              _Float16* __restrict__ out) {
  const int kt = K >> 5, nt = ncols >> 4;
  const int total = kt * nt * 32;
  const int t = blockIdx.x * blockDim.x + threadIdx.x;
  if (t >= total) return;
  const int lane = t & 31;
  const int frag = t >> 5;
  const int ni = frag % nt, ki = frag / nt;
  const int n = lane & 15, hi = lane >> 4;
  const float* p = W + (size_t)(ki * 32 + hi * 8) * ncols + ni * 16 + n;
  _Float16* o = out + (size_t)frag * FRAGH + lane * 16;
#pragma unroll
  for (int j = 0; j < 16; ++j) {
    const int k = ((j >> 3) << 4) + (j & 7);
    o[j] = (_Float16)p[(size_t)k * ncols];
  }
}

// ---- Input projection: X0[N,64] = [xc@Wp+bp | xs@Ws+bs] --------------------
// 8 waves/block; wave -> one 16x16 tile; block covers 32 rows x 64 cols.
__global__ __launch_bounds__(256) void proj_kernel(
    const float* __restrict__ xc, const float* __restrict__ xs,
    const _Float16* __restrict__ PWp, const float* __restrict__ bp,
    const _Float16* __restrict__ PWs, const float* __restrict__ bs,
    float* __restrict__ X0) {
  const int lane = threadIdx.x & 31;
  const int wave = threadIdx.x >> 5;
  const int rbase = blockIdx.x * 32 + (wave >> 2) * 16;
  if (rbase >= NNODES) return;                  // uniform per wave
  const int col_tile = wave & 3;                // 0,1: content  2,3: style
  const bool style = (col_tile >= 2);
  const float*    Xsrc = style ? xs : xc;
  const _Float16* P    = style ? PWs : PWp;
  const float*    bias = style ? bs : bp;
  const int       K    = style ? 128 : 768;
  const int       ni   = col_tile & 1;          // weight column tile (nt = 2)

  v8f acc = {};
  for (int k = 0; k < K; k += 32) {
    v16h a = load_a_frag(Xsrc, K, rbase, k, lane);
    v16h b = load_b_packed(P, 2, k >> 5, ni, lane);
    acc = wmma_f16(a, b, acc);
  }

  const int n  = lane & 15;
  const int hi = lane >> 4;
  const float bv = bias[ni * 16 + n];
  float* orow = X0 + (size_t)(rbase + hi * 8) * DFEAT + (style ? 32 : 0) + ni * 16 + n;
#pragma unroll
  for (int r = 0; r < 8; ++r) orow[(size_t)r * DFEAT] = acc[r] + bv;
}

// ---- SAGE layer: xout = relu( (msg*invdeg)@Wl + bl + xin@Wr ) --------------
__global__ __launch_bounds__(256) void sage_kernel(
    const float* __restrict__ msg, const float* __restrict__ invdeg,
    const float* __restrict__ xin,
    const _Float16* __restrict__ PWl, const float* __restrict__ bl,
    const _Float16* __restrict__ PWr, float* __restrict__ xout) {
  const int lane = threadIdx.x & 31;
  const int wave = threadIdx.x >> 5;
  const int rbase = blockIdx.x * 32 + (wave >> 2) * 16;
  if (rbase >= NNODES) return;
  const int ni = wave & 3;                      // column tile (nt = 4)

  v8f acc = {};
#pragma unroll
  for (int k = 0; k < DFEAT; k += 32) {
    v16h aa = load_a_frag_scaled(msg, DFEAT, rbase, k, lane, invdeg);
    v16h ba = load_b_packed(PWl, 4, k >> 5, ni, lane);
    acc = wmma_f16(aa, ba, acc);
    v16h ax = load_a_frag(xin, DFEAT, rbase, k, lane);
    v16h bx = load_b_packed(PWr, 4, k >> 5, ni, lane);
    acc = wmma_f16(ax, bx, acc);
  }

  const int n  = lane & 15;
  const int hi = lane >> 4;
  const float bv = bl[ni * 16 + n];
  float* orow = xout + (size_t)(rbase + hi * 8) * DFEAT + ni * 16 + n;
#pragma unroll
  for (int r = 0; r < 8; ++r) {
    const float v = acc[r] + bv;
    orow[(size_t)r * DFEAT] = v > 0.0f ? v : 0.0f;
  }
}

// ---- Graph aggregation helpers --------------------------------------------
__global__ void fill_kernel(float* __restrict__ p, int n, float v) {
  for (int i = blockIdx.x * blockDim.x + threadIdx.x; i < n;
       i += gridDim.x * blockDim.x)
    p[i] = v;
}

__global__ void fill4_kernel(float4* __restrict__ p, int n4) {
  const float4 z = make_float4(0.f, 0.f, 0.f, 0.f);
  for (int i = blockIdx.x * blockDim.x + threadIdx.x; i < n4;
       i += gridDim.x * blockDim.x)
    p[i] = z;
}

__global__ void degree_kernel(const int* __restrict__ dst,
                              float* __restrict__ deg, int nedges) {
  for (int e = blockIdx.x * blockDim.x + threadIdx.x; e < nedges;
       e += gridDim.x * blockDim.x)
    atomicAdd(&deg[dst[e]], 1.0f);
}

__global__ void invdeg_kernel(float* __restrict__ deg, int n) {
  for (int i = blockIdx.x * blockDim.x + threadIdx.x; i < n;
       i += gridDim.x * blockDim.x)
    deg[i] = 1.0f / fmaxf(deg[i], 1.0f);
}

// 64 consecutive lanes handle one edge: each atomic instruction covers one
// contiguous 128B destination row segment (half of a 256B row) (L2-friendly).
__global__ void scatter_kernel(const float* __restrict__ x,
                               const int* __restrict__ src,
                               const int* __restrict__ dst,
                               float* __restrict__ msg, int total) {
  for (int i = blockIdx.x * blockDim.x + threadIdx.x; i < total;
       i += gridDim.x * blockDim.x) {
    const int e = i >> 6;
    const int f = i & 63;
    const int s = src[e];
    const int d = dst[e];
    atomicAdd(&msg[(size_t)d * DFEAT + f], x[(size_t)s * DFEAT + f]);
  }
}

// ---- Head: out[N,2] = x@Wo + bo -------------------------------------------
__global__ void head_kernel(const float* __restrict__ x,
                            const float* __restrict__ Wo,
                            const float* __restrict__ bo,
                            float* __restrict__ out) {
  const int i = blockIdx.x * blockDim.x + threadIdx.x;
  if (i >= NNODES) return;
  const float* row = x + (size_t)i * DFEAT;
  float o0 = bo[0], o1 = bo[1];
#pragma unroll
  for (int k = 0; k < DFEAT; ++k) {
    const float v = row[k];
    o0 += v * Wo[k * 2 + 0];
    o1 += v * Wo[k * 2 + 1];
  }
  out[i * 2 + 0] = o0;
  out[i * 2 + 1] = o1;
}

// ---------------------------------------------------------------------------
extern "C" void kernel_launch(void* const* d_in, const int* in_sizes, int n_in,
                              void* d_out, int out_size, void* d_ws,
                              size_t ws_size, hipStream_t stream) {
  const float* xc  = (const float*)d_in[0];    // [N,768]
  const float* xs  = (const float*)d_in[1];    // [N,128]
  const int*   ei  = (const int*)d_in[2];      // [2,E] flat
  /* edge_type d_in[3] unused (eval path) */
  const float* Wp  = (const float*)d_in[4];
  const float* bp  = (const float*)d_in[5];
  const float* Ws  = (const float*)d_in[6];
  const float* bs  = (const float*)d_in[7];
  const float* Wl1 = (const float*)d_in[8];
  const float* bl1 = (const float*)d_in[9];
  const float* Wr1 = (const float*)d_in[10];
  const float* Wl2 = (const float*)d_in[11];
  const float* bl2 = (const float*)d_in[12];
  const float* Wr2 = (const float*)d_in[13];
  const float* Wl3 = (const float*)d_in[14];
  const float* bl3 = (const float*)d_in[15];
  const float* Wr3 = (const float*)d_in[16];
  const float* Wo  = (const float*)d_in[17];
  const float* bo  = (const float*)d_in[18];

  const int E = in_sizes[2] / 2;
  const int* src = ei;
  const int* dst = ei + E;

  float* ws  = (float*)d_ws;
  float* X0  = ws;                              // [N,64]
  float* X1  = ws + (size_t)NNODES * DFEAT;     // [N,64]
  float* MSG = ws + 2ull * NNODES * DFEAT;      // [N,64]
  float* IDG = ws + 3ull * NNODES * DFEAT;      // [N]
  // Packed f16 weight fragments (32B aligned: 193*N*4 bytes is 32-divisible).
  _Float16* PW   = (_Float16*)(ws + 3ull * NNODES * DFEAT + NNODES);
  _Float16* PWp  = PW;                          // 24*2 = 48 frags
  _Float16* PWs  = PW + 48ull  * FRAGH;         //  4*2 =  8 frags
  _Float16* PWl1 = PW + 56ull  * FRAGH;         //  2*4 =  8 frags each below
  _Float16* PWr1 = PW + 64ull  * FRAGH;
  _Float16* PWl2 = PW + 72ull  * FRAGH;
  _Float16* PWr2 = PW + 80ull  * FRAGH;
  _Float16* PWl3 = PW + 88ull  * FRAGH;
  _Float16* PWr3 = PW + 96ull  * FRAGH;

  const int gemm_blocks = (NNODES + 31) / 32;   // 32 rows / block
  const int feat_elems  = NNODES * DFEAT;
  const int scat_total  = E * DFEAT;

  // Prepack weights into WMMA B-fragments (tiny one-shot kernels).
  pack_b_kernel<<<24, 64, 0, stream>>>(Wp,  768, 32, PWp);   // 1536 thr
  pack_b_kernel<<<4,  64, 0, stream>>>(Ws,  128, 32, PWs);   //  256 thr
  pack_b_kernel<<<4,  64, 0, stream>>>(Wl1, 64,  64, PWl1);  //  256 thr
  pack_b_kernel<<<4,  64, 0, stream>>>(Wr1, 64,  64, PWr1);
  pack_b_kernel<<<4,  64, 0, stream>>>(Wl2, 64,  64, PWl2);
  pack_b_kernel<<<4,  64, 0, stream>>>(Wr2, 64,  64, PWr2);
  pack_b_kernel<<<4,  64, 0, stream>>>(Wl3, 64,  64, PWl3);
  pack_b_kernel<<<4,  64, 0, stream>>>(Wr3, 64,  64, PWr3);

  // Degree (layer-invariant): deg -> 1/max(deg,1)
  fill_kernel<<<256, 256, 0, stream>>>(IDG, NNODES, 0.0f);
  degree_kernel<<<(E + 255) / 256, 256, 0, stream>>>(dst, IDG, E);
  invdeg_kernel<<<(NNODES + 255) / 256, 256, 0, stream>>>(IDG, NNODES);

  // Input projection
  proj_kernel<<<gemm_blocks, 256, 0, stream>>>(xc, xs, PWp, bp, PWs, bs, X0);

  // Layer 1: X0 -> X1
  fill4_kernel<<<2048, 256, 0, stream>>>((float4*)MSG, feat_elems / 4);
  scatter_kernel<<<16384, 256, 0, stream>>>(X0, src, dst, MSG, scat_total);
  sage_kernel<<<gemm_blocks, 256, 0, stream>>>(MSG, IDG, X0, PWl1, bl1, PWr1, X1);

  // Layer 2: X1 -> X0
  fill4_kernel<<<2048, 256, 0, stream>>>((float4*)MSG, feat_elems / 4);
  scatter_kernel<<<16384, 256, 0, stream>>>(X1, src, dst, MSG, scat_total);
  sage_kernel<<<gemm_blocks, 256, 0, stream>>>(MSG, IDG, X1, PWl2, bl2, PWr2, X0);

  // Layer 3: X0 -> X1
  fill4_kernel<<<2048, 256, 0, stream>>>((float4*)MSG, feat_elems / 4);
  scatter_kernel<<<16384, 256, 0, stream>>>(X0, src, dst, MSG, scat_total);
  sage_kernel<<<gemm_blocks, 256, 0, stream>>>(MSG, IDG, X0, PWl3, bl3, PWr3, X1);

  // Classifier head
  head_kernel<<<(NNODES + 255) / 256, 256, 0, stream>>>(X1, Wo, bo, (float*)d_out);
}
